// RecommendationModel_17428977287550
// MI455X (gfx1250) — matrix operand order repaired and checked by reading, not compiled, not run on hardware
//
#include <hip/hip_runtime.h>
#include <hip/hip_bf16.h>
#include <math.h>

// ---------------- model constants ----------------
#define H      128
#define NUSER  8192
#define NREC   8192
#define FIN    20
#define ETASTE 1000000
#define EBUY   500000
// TEMP = 0.5  -> logits scaled by 2.0
#define INV_SQRT_H 0.08838834764831845f   // 1/sqrt(128)

typedef __attribute__((ext_vector_type(16))) __bf16 v16bf;
typedef __attribute__((ext_vector_type(8)))  float  v8f;
typedef unsigned int u32x4 __attribute__((ext_vector_type(4)));
typedef int          i32x4 __attribute__((ext_vector_type(4)));
typedef int          i32x8 __attribute__((ext_vector_type(8)));

enum { EPI_NONE = 0, EPI_BIAS = 1, EPI_BIAS_RELU = 2, EPI_SKIP = 3 };

#if __has_builtin(__builtin_amdgcn_tensor_load_to_lds)
#define USE_TDM 1
#endif

// ---------------- helpers ----------------
__device__ __forceinline__ float wave_sum(float v) {
#pragma unroll
  for (int m = 1; m < 32; m <<= 1) v += __shfl_xor(v, m, 32);
  return v;
}
__device__ __forceinline__ float half16_sum(float v) {
#pragma unroll
  for (int m = 1; m < 16; m <<= 1) v += __shfl_xor(v, m, 32);  // stays inside 16-lane half
  return v;
}
// monotonic float <-> uint mapping for atomicMax-based segment max
__device__ __forceinline__ unsigned fenc(float f) {
  unsigned u = __float_as_uint(f);
  return (u >> 31) ? ~u : (u | 0x80000000u);
}
__device__ __forceinline__ float fdec(unsigned e) {
  return (e >> 31) ? __uint_as_float(e ^ 0x80000000u) : __uint_as_float(~e);
}

#ifdef USE_TDM
// TDM: DMA a rows x cols f32 tile (row stride strideElems) from global into LDS
// at byte offset ldsOff, packed row-major (cols elements per row).
// D# group0: [count=1 | lds_addr | global_addr(57b) | type=2]
// D# group1: [data_size=4B | tensor dims (oversized: tile never OOB) |
//             tile_dim0=cols, tile_dim1=rows | tensor_dim0_stride=strideElems]
__device__ __forceinline__ void tdm_load_tile_f32(unsigned ldsOff, const float* gsrc,
                                                  int rows, int cols, int strideElems) {
  unsigned long long ga = (unsigned long long)(size_t)gsrc;
  u32x4 g0;
  g0[0] = 1u;                                            // count=1, user mode
  g0[1] = ldsOff;                                        // lds_addr (bytes)
  g0[2] = (unsigned)ga;                                  // global_addr[31:0]
  g0[3] = (unsigned)((ga >> 32) & 0x1FFFFFFu) | (2u << 30);  // addr[56:32] | type=2
  const unsigned TD = 1u << 20;                          // huge tensor dims -> no OOB
  i32x8 g1;
  g1[0] = 0x00020000;                                    // data_size = 2 (4 bytes)
  g1[1] = (int)((TD & 0xFFFFu) << 16);                   // tensor_dim0[15:0]
  g1[2] = (int)(((TD >> 16) & 0xFFFFu) | ((TD & 0xFFFFu) << 16));   // td0[31:16]|td1[15:0]
  g1[3] = (int)(((TD >> 16) & 0xFFFFu) | ((unsigned)cols << 16));   // td1[31:16]|tile_dim0
  g1[4] = rows;                                          // tile_dim1 (tile_dim2 = 0)
  g1[5] = strideElems;                                   // tensor_dim0_stride[31:0]
  g1[6] = 0;                                             // stride0[47:32] | stride1[15:0]
  g1[7] = 0;
  i32x4 z4 = {0, 0, 0, 0};
#if __clang_major__ >= 23
  i32x8 z8 = {0, 0, 0, 0, 0, 0, 0, 0};
  __builtin_amdgcn_tensor_load_to_lds(g0, g1, z4, z4, z8, 0);
#else
  __builtin_amdgcn_tensor_load_to_lds(g0, g1, z4, z4, 0);
#endif
}
#endif

// ---------------- generic bf16-WMMA GEMM (templated epilogue) ----------------
// grid = (M/16, (N/16)/8); 8 waves/block, wave w owns tile (blockIdx.x, by*8+w).
// A tile (16 x K f32) is TDM-staged into LDS (overlapped with B staging below).
// B (KxN f32, row stride ldb) 128-column slice is staged transposed+bf16 into LDS,
// so each wave's B fragment is one aligned 32B LDS vector load.
template <int EPI>
__global__ void k_gemm(const float* __restrict__ A, int lda,
                       const float* __restrict__ B, int ldb,
                       const float* __restrict__ bias,
                       float* __restrict__ C, int ldc,
                       const float* __restrict__ Xin,
                       const float* __restrict__ skipArr, int skipIdx,
                       int M, int N, int K) {
  __shared__ __align__(16) float  tA[16 * 128];          // 8 KB
  __shared__ __align__(32) __bf16 tB[128 * 128];         // 32 KB, [c][k] c=local col
  int mt = blockIdx.x;
  int w  = threadIdx.x >> 5;                             // wave id = local col tile
  int nt = blockIdx.y * 8 + w;
  int cbase = blockIdx.y * 128;

  // -- stage A via TDM (wave0), overlap with cooperative B transpose+convert --
#ifdef USE_TDM
  if (threadIdx.x < 32)
    tdm_load_tile_f32((unsigned)(size_t)tA, A + (size_t)mt * 16 * lda, 16, K, lda);
#else
  for (int i = threadIdx.x; i < 16 * K; i += blockDim.x)
    tA[i] = A[(size_t)(mt * 16 + i / K) * lda + (i % K)];
#endif
  for (int i = threadIdx.x; i < K * 128; i += blockDim.x) {
    int k = i >> 7, c = i & 127;                         // coalesced read of B row k
    tB[c * K + k] = (__bf16)B[(size_t)k * ldb + cbase + c];
  }
#ifdef USE_TDM
  if (threadIdx.x < 32) __builtin_amdgcn_s_wait_tensorcnt(0);
#endif
  __syncthreads();

  int lane = threadIdx.x & 31;
  int half = lane >> 4, mn = lane & 15;
  const float*  lrA = tA + mn * K;                       // A row for this lane
  const __bf16* lcB = tB + (w * 16 + mn) * K;            // B column for this lane
  v8f acc = {0.f, 0.f, 0.f, 0.f, 0.f, 0.f, 0.f, 0.f};

  for (int k0 = 0; k0 < K; k0 += 32) {
    v16bf a;
    const float* pa = lrA + k0 + 16 * half;
#pragma unroll
    for (int j = 0; j < 8; j++) { a[j] = (__bf16)pa[j]; a[8 + j] = (__bf16)pa[8 + j]; }
    v16bf b = *(const v16bf*)(lcB + k0 + 16 * half);     // 32B aligned LDS load
    acc = __builtin_amdgcn_wmma_f32_16x16x32_bf16(false, a, false, b,
                                                  (short)0, acc, false, false);
  }

  int gn = nt * 16 + mn;
  float bv = (EPI != EPI_NONE) ? bias[gn] : 0.f;
  float g = 0.f;
  if (EPI == EPI_SKIP) g = 1.f / (1.f + __expf(-skipArr[skipIdx]));
#pragma unroll
  for (int r = 0; r < 8; r++) {                          // D: row = 8*half + r, col = mn
    int gm = mt * 16 + 8 * half + r;
    float v = acc[r] + bv;
    if (EPI == EPI_BIAS_RELU)     v = fmaxf(v, 0.f);
    else if (EPI == EPI_SKIP)     v = g * v + (1.f - g) * Xin[(size_t)gm * ldc + gn];
    C[(size_t)gm * ldc + gn] = v;
  }
}

// ---------------- fused sim GEMM: rowsum(exp) + diagonal, 256MB never materialized ----
// grid = (NREC/16, (NREC/16)/8); z1 tile TDM-staged, z2 rows read directly (contiguous).
__global__ void k_sim(const float* __restrict__ z1, const float* __restrict__ z2,
                      float* __restrict__ rowsum, float* __restrict__ diag) {
  __shared__ __align__(16) float tA[16 * H];
  int mt = blockIdx.x;
  int nt = blockIdx.y * 8 + (threadIdx.x >> 5);

#ifdef USE_TDM
  if (threadIdx.x < 32) {
    tdm_load_tile_f32((unsigned)(size_t)tA, z1 + (size_t)mt * 16 * H, 16, H, H);
    __builtin_amdgcn_s_wait_tensorcnt(0);
  }
#else
  for (int i = threadIdx.x; i < 16 * H; i += blockDim.x)
    tA[i] = z1[(size_t)mt * 16 * H + i];
#endif
  __syncthreads();

  int lane = threadIdx.x & 31;
  int half = lane >> 4, mn = lane & 15;
  const float* lr   = tA + mn * H;
  const float* Brow = z2 + (size_t)(nt * 16 + mn) * H;   // z2 rows == Bt
  v8f acc = {0.f, 0.f, 0.f, 0.f, 0.f, 0.f, 0.f, 0.f};
  for (int k0 = 0; k0 < H; k0 += 32) {
    v16bf a, b;
    const float* pa = lr + k0 + 16 * half;
#pragma unroll
    for (int j = 0; j < 8; j++) { a[j] = (__bf16)pa[j]; a[8 + j] = (__bf16)pa[8 + j]; }
    const float* pb = Brow + k0 + 16 * half;
#pragma unroll
    for (int e = 0; e < 16; e++) b[e] = (__bf16)pb[e];
    acc = __builtin_amdgcn_wmma_f32_16x16x32_bf16(false, a, false, b,
                                                  (short)0, acc, false, false);
  }
#pragma unroll
  for (int r = 0; r < 8; r++) {
    float s = __expf(2.f * acc[r]);          // sim = dot/TEMP, TEMP=0.5
    s = half16_sum(s);                       // sum over 16 cols of this tile row
    if (mn == 0) atomicAdd(&rowsum[mt * 16 + 8 * half + r], s);
  }
  if (mt == nt) {
#pragma unroll
    for (int r = 0; r < 8; r++) {
      int m = 8 * half + r;
      if (mn == m) diag[mt * 16 + m] = 2.f * acc[r];
    }
  }
}

__global__ void k_clloss(const float* __restrict__ diag,
                         const float* __restrict__ rowsum, float* __restrict__ cl) {
  __shared__ float sm[256];
  float p = 0.f;
  for (int i = threadIdx.x; i < NREC; i += 256) p += diag[i] - __logf(rowsum[i]);
  sm[threadIdx.x] = p;
  __syncthreads();
  for (int s = 128; s > 0; s >>= 1) {
    if ((int)threadIdx.x < s) sm[threadIdx.x] += sm[threadIdx.x + s];
    __syncthreads();
  }
  if (threadIdx.x == 0) cl[0] = -sm[0] / (float)NREC;
}

// ---------------- gathers / elementwise ----------------
__global__ void k_gather(float* __restrict__ dst, const float* __restrict__ tbl,
                         const int* __restrict__ ids, int rows) {
  int row = blockIdx.x * (blockDim.x >> 5) + (threadIdx.x >> 5);
  if (row >= rows) return;
  int lane = threadIdx.x & 31;
  ((float4*)(dst + (size_t)row * H))[lane] =
      ((const float4*)(tbl + (size_t)ids[row] * H))[lane];
}

__global__ void k_user_renorm(float* __restrict__ dst, const float* __restrict__ emb,
                              const int* __restrict__ ids, int rows) {
  int row = blockIdx.x * (blockDim.x >> 5) + (threadIdx.x >> 5);
  if (row >= rows) return;
  int lane = threadIdx.x & 31;
  float4 v = ((const float4*)(emb + (size_t)ids[row] * H))[lane];
  float ss = wave_sum(v.x * v.x + v.y * v.y + v.z * v.z + v.w * v.w);
  float n = sqrtf(ss);
  float sc = (n > 1.f) ? 1.f / (n + 1e-7f) : 1.f;
  v.x *= sc; v.y *= sc; v.z *= sc; v.w *= sc;
  ((float4*)(dst + (size_t)row * H))[lane] = v;
}

__global__ void k_rownorm(float* __restrict__ z, int rows) {
  int row = blockIdx.x * (blockDim.x >> 5) + (threadIdx.x >> 5);
  if (row >= rows) return;
  int lane = threadIdx.x & 31;
  float4* p = (float4*)(z + (size_t)row * H);
  float4 v = p[lane];
  float ss = wave_sum(v.x * v.x + v.y * v.y + v.z * v.z + v.w * v.w);
  float inv = 1.f / fmaxf(sqrtf(ss), 1e-12f);
  v.x *= inv; v.y *= inv; v.z *= inv; v.w *= inv;
  p[lane] = v;
}

__global__ void k_nutrient(float* __restrict__ dst, const float* __restrict__ nut,
                           const float* __restrict__ W, const float* __restrict__ b) {
  int i = blockIdx.x * blockDim.x + threadIdx.x;
  if (i >= NREC * H) return;
  int r = i >> 7, c = i & 127;
  float s = b[c];
#pragma unroll
  for (int f = 0; f < FIN; f++) s += nut[r * FIN + f] * W[f * H + c];
  dst[i] = s;
}

__global__ void k_qcopy(float* __restrict__ dst, const float* __restrict__ kqv, int rows) {
  int i = blockIdx.x * blockDim.x + threadIdx.x;
  if (i >= rows * H) return;
  int r = i >> 7, c = i & 127;
  dst[i] = kqv[(size_t)r * 384 + 128 + c];
}

__global__ void k_gelu(float* __restrict__ x, int n) {
  int i = blockIdx.x * blockDim.x + threadIdx.x;
  if (i >= n) return;
  float v = x[i];
  x[i] = 0.5f * v * (1.f + erff(v * 0.70710678118654752f));   // exact gelu
}

// ---------------- TasteGNN (LightGCN) ----------------
__global__ void k_deg(float* __restrict__ deg, const int* __restrict__ col, int E) {
  int i = blockIdx.x * blockDim.x + threadIdx.x;
  if (i < E) atomicAdd(deg + col[i], 1.f);
}
__global__ void k_dinv(float* __restrict__ deg, int n) {
  int i = blockIdx.x * blockDim.x + threadIdx.x;
  if (i >= n) return;
  float d = deg[i];
  deg[i] = (d > 0.f) ? 1.f / sqrtf(d) : 0.f;
}
__global__ void k_taste_scatter(float* __restrict__ taste, const float* __restrict__ cook,
                                const float* __restrict__ dinv,
                                const int* __restrict__ row, const int* __restrict__ col, int E) {
  int e = blockIdx.x * (blockDim.x >> 5) + (threadIdx.x >> 5);
  if (e >= E) return;
  int lane = threadIdx.x & 31;
  int r = row[e], c = col[e];
  float w = dinv[r] * dinv[c];
  float4 v = ((const float4*)(cook + (size_t)r * H))[lane];
  float* o = taste + (size_t)c * H + lane * 4;
  atomicAdd(o + 0, w * v.x); atomicAdd(o + 1, w * v.y);
  atomicAdd(o + 2, w * v.z); atomicAdd(o + 3, w * v.w);
}

// ---------------- HGT edge softmax ----------------
__global__ void k_logit(float* __restrict__ lbuf, unsigned* __restrict__ mbuf,
                        const float* __restrict__ q, const float* __restrict__ ke,
                        const int* __restrict__ src, const int* __restrict__ dst, int E,
                        const float* __restrict__ p_rel, int eidx) {
  int e = blockIdx.x * (blockDim.x >> 5) + (threadIdx.x >> 5);
  if (e >= E) return;
  int lane = threadIdx.x & 31;
  int s = src[e], d = dst[e];
  float4 qv = ((const float4*)(q + (size_t)d * H))[lane];
  float4 kv = ((const float4*)(ke + (size_t)s * H))[lane];
  float p = wave_sum(qv.x * kv.x + qv.y * kv.y + qv.z * kv.z + qv.w * kv.w);
  if (lane == 0) {
    float logit = p * p_rel[eidx] * INV_SQRT_H;
    lbuf[e] = logit;
    atomicMax(mbuf + d, fenc(logit));
  }
}
__global__ void k_pass2(float* __restrict__ lbuf, const unsigned* __restrict__ mbuf,
                        float* __restrict__ sbuf, const int* __restrict__ dst, int E) {
  int i = blockIdx.x * blockDim.x + threadIdx.x;
  if (i >= E) return;
  int d = dst[i];
  float e = __expf(lbuf[i] - fdec(mbuf[d]));
  lbuf[i] = e;
  atomicAdd(sbuf + d, e);
}
__global__ void k_pass3(float* __restrict__ agg, const float* __restrict__ lbuf,
                        const float* __restrict__ sbuf, const float* __restrict__ ve,
                        const int* __restrict__ src, const int* __restrict__ dst, int E) {
  int e = blockIdx.x * (blockDim.x >> 5) + (threadIdx.x >> 5);
  if (e >= E) return;
  int lane = threadIdx.x & 31;
  int s = src[e], d = dst[e];
  float alpha = lbuf[e] / (sbuf[d] + 1e-16f);
  float4 v = ((const float4*)(ve + (size_t)s * H))[lane];
  float* o = agg + (size_t)d * H + lane * 4;
  atomicAdd(o + 0, alpha * v.x); atomicAdd(o + 1, alpha * v.y);
  atomicAdd(o + 2, alpha * v.z); atomicAdd(o + 3, alpha * v.w);
}

// ---------------- final assembly: += cl_loss, copy taste section ----------------
__global__ void k_final(float* __restrict__ out, const float* __restrict__ taste,
                        const float* __restrict__ cl) {
  int i = blockIdx.x * blockDim.x + threadIdx.x;
  if (i >= (NUSER + 2 * NREC) * H) return;
  float c = cl[0];
  if (i < (NUSER + NREC) * H) out[i] += c;
  else                        out[i] = taste[i - (NUSER + NREC) * H] + c;
}

// ---------------- host ----------------
static void gemm(hipStream_t st, const float* A, int lda, const float* B, int ldb,
                 const float* bias, float* C, int ldc, const float* Xin,
                 const float* skipArr, int skipIdx, int M, int N, int K, int epi) {
  dim3 grid(M / 16, (N / 16) / 8);
  switch (epi) {
    case EPI_NONE:
      k_gemm<EPI_NONE><<<grid, 256, 0, st>>>(A, lda, B, ldb, bias, C, ldc, Xin,
                                             skipArr, skipIdx, M, N, K); break;
    case EPI_BIAS:
      k_gemm<EPI_BIAS><<<grid, 256, 0, st>>>(A, lda, B, ldb, bias, C, ldc, Xin,
                                             skipArr, skipIdx, M, N, K); break;
    case EPI_BIAS_RELU:
      k_gemm<EPI_BIAS_RELU><<<grid, 256, 0, st>>>(A, lda, B, ldb, bias, C, ldc, Xin,
                                                  skipArr, skipIdx, M, N, K); break;
    default:
      k_gemm<EPI_SKIP><<<grid, 256, 0, st>>>(A, lda, B, ldb, bias, C, ldc, Xin,
                                             skipArr, skipIdx, M, N, K); break;
  }
}

extern "C" void kernel_launch(void* const* d_in, const int* in_sizes, int n_in,
                              void* d_out, int out_size, void* d_ws, size_t ws_size,
                              hipStream_t stream) {
  (void)in_sizes; (void)n_in; (void)out_size; (void)ws_size;
  const float* user_emb      = (const float*)d_in[0];
  const float* visual_table  = (const float*)d_in[1];
  const float* caption_table = (const float*)d_in[2];
  const float* cooking_table = (const float*)d_in[3];
  const float* nutrient      = (const float*)d_in[5];
  const float* item_x_in     = (const float*)d_in[6];
  const float* W_nut = (const float*)d_in[7];
  const float* b_nut = (const float*)d_in[8];
  const float* W_fc1 = (const float*)d_in[9];
  const float* b_fc1 = (const float*)d_in[10];
  const float* W_fc2 = (const float*)d_in[11];
  const float* b_fc2 = (const float*)d_in[12];
  const float* W_kqv = (const float*)d_in[13];
  const float* b_kqv = (const float*)d_in[14];
  const float* W_out = (const float*)d_in[15];
  const float* b_out = (const float*)d_in[16];
  const float* skipA = (const float*)d_in[17];
  const float* a_rel = (const float*)d_in[18];
  const float* m_rel = (const float*)d_in[19];
  const float* p_rel = (const float*)d_in[20];
  const int* user_ids  = (const int*)d_in[21];
  const int* image_ids = (const int*)d_in[22];
  const int* taste_ids = (const int*)d_in[24];
  const int* tei       = (const int*)d_in[25];
  const int* e_ti      = (const int*)d_in[26];   // taste->item
  const int* e_ii      = (const int*)d_in[27];   // intention->item
  const int* e_gi      = (const int*)d_in[28];   // image->item
  const int* e_ui      = (const int*)d_in[29];   // user->item
  const int* e_iu      = (const int*)d_in[30];   // item->user
  float* out = (float*)d_out;

  // ---- workspace layout (~110 MB) ----
  float* ws = (float*)d_ws;
  size_t off = 0;
  auto alloc = [&](size_t n) { float* p = ws + off; off += (n + 63) & ~(size_t)63; return p; };
  float* user_x    = alloc((size_t)NUSER * H);
  float* caption_x = alloc((size_t)NREC * H);
  float* visual_x  = alloc((size_t)NREC * H);
  float* cooking_x = alloc((size_t)NREC * H);
  float* nutri_x   = alloc((size_t)NREC * H);
  float* htmp      = alloc((size_t)NREC * H);
  float* z1        = alloc((size_t)NREC * H);
  float* z2        = alloc((size_t)NREC * H);
  float* taste_x   = alloc((size_t)NREC * H);
  float* kqv_tmp   = alloc((size_t)NREC * 384);
  float* q_user    = alloc((size_t)NUSER * H);
  float* q_item    = alloc((size_t)NREC * H);
  float* keB[5]; for (int i = 0; i < 5; i++) keB[i] = alloc((size_t)NREC * H);
  float* veB[5]; for (int i = 0; i < 5; i++) veB[i] = alloc((size_t)NREC * H);
  float* agg_item  = alloc((size_t)NREC * H);
  float* agg_user  = alloc((size_t)NUSER * H);
  float* lb[5];
  lb[0] = alloc(NREC); lb[1] = alloc(NREC); lb[2] = alloc(NREC);
  lb[3] = alloc(EBUY); lb[4] = alloc(EBUY);
  float* rowsum = alloc(NREC);
  float* diag   = alloc(NREC);
  float* deg    = alloc(NREC);
  float* m_item = alloc(NREC);
  float* s_item = alloc(NREC);
  float* m_user = alloc(NUSER);
  float* s_user = alloc(NUSER);
  float* cl     = alloc(64);

  // ---- zero accumulators (memset-0 is valid identity for fenc-atomicMax too) ----
  hipMemsetAsync(rowsum,  0, NREC  * 4, stream);
  hipMemsetAsync(deg,     0, NREC  * 4, stream);
  hipMemsetAsync(taste_x, 0, (size_t)NREC * H * 4, stream);
  hipMemsetAsync(m_item,  0, NREC  * 4, stream);
  hipMemsetAsync(s_item,  0, NREC  * 4, stream);
  hipMemsetAsync(m_user,  0, NUSER * 4, stream);
  hipMemsetAsync(s_user,  0, NUSER * 4, stream);
  hipMemsetAsync(agg_item, 0, (size_t)NREC  * H * 4, stream);
  hipMemsetAsync(agg_user, 0, (size_t)NUSER * H * 4, stream);

  // ---- encoders / gathers ----
  k_user_renorm<<<NUSER / 8, 256, 0, stream>>>(user_x, user_emb, user_ids, NUSER);
  k_gather<<<NREC / 8, 256, 0, stream>>>(visual_x,  visual_table,  image_ids, NREC);
  k_gather<<<NREC / 8, 256, 0, stream>>>(caption_x, caption_table, image_ids, NREC);
  k_gather<<<NREC / 8, 256, 0, stream>>>(cooking_x, cooking_table, taste_ids, NREC);
  k_nutrient<<<(NREC * H) / 256, 256, 0, stream>>>(nutri_x, nutrient, W_nut, b_nut);

  // ---- shared contrastive encoder: z = normalize(relu(x@W1+b1)@W2+b2) ----
  gemm(stream, nutri_x, H, W_fc1, 128, b_fc1, htmp, 128, nullptr, nullptr, 0,
       NREC, 128, H, EPI_BIAS_RELU);
  gemm(stream, htmp, 128, W_fc2, H, b_fc2, z1, H, nullptr, nullptr, 0,
       NREC, H, 128, EPI_BIAS);
  k_rownorm<<<NREC / 8, 256, 0, stream>>>(z1, NREC);
  gemm(stream, caption_x, H, W_fc1, 128, b_fc1, htmp, 128, nullptr, nullptr, 0,
       NREC, 128, H, EPI_BIAS_RELU);
  gemm(stream, htmp, 128, W_fc2, H, b_fc2, z2, H, nullptr, nullptr, 0,
       NREC, H, 128, EPI_BIAS);
  k_rownorm<<<NREC / 8, 256, 0, stream>>>(z2, NREC);

  // ---- InfoNCE: fused sim GEMM + logsumexp/diag, then scalar loss ----
  k_sim<<<dim3(NREC / 16, (NREC / 16) / 8), 256, 0, stream>>>(z1, z2, rowsum, diag);
  k_clloss<<<1, 256, 0, stream>>>(diag, rowsum, cl);

  // ---- TasteGNN (LightGCN over 1M edges) ----
  k_deg<<<(ETASTE + 255) / 256, 256, 0, stream>>>(deg, tei + ETASTE, ETASTE);
  k_dinv<<<NREC / 256, 256, 0, stream>>>(deg, NREC);
  k_taste_scatter<<<ETASTE / 8, 256, 0, stream>>>(taste_x, cooking_x, deg,
                                                  tei, tei + ETASTE, ETASTE);

  // ---- HGT: kqv per node type, per-edge-type key/value transforms ----
  const float* xs[5] = {user_x, item_x_in, taste_x, z2, visual_x};
  const int e_of_t[5] = {3, 4, 0, 1, 2};    // node type t -> its outgoing edge type
  for (int t = 0; t < 5; t++) {
    gemm(stream, xs[t], H, W_kqv + (size_t)t * H * 384, 384, b_kqv + (size_t)t * 384,
         kqv_tmp, 384, nullptr, nullptr, 0, NREC, 384, H, EPI_BIAS);
    if (t == 0) k_qcopy<<<(NUSER * H) / 256, 256, 0, stream>>>(q_user, kqv_tmp, NUSER);
    if (t == 1) k_qcopy<<<(NREC * H) / 256, 256, 0, stream>>>(q_item, kqv_tmp, NREC);
    int e = e_of_t[t];
    gemm(stream, kqv_tmp + 0,   384, a_rel + (size_t)e * H * H, H, nullptr,
         keB[e], H, nullptr, nullptr, 0, NREC, H, H, EPI_NONE);
    gemm(stream, kqv_tmp + 256, 384, m_rel + (size_t)e * H * H, H, nullptr,
         veB[e], H, nullptr, nullptr, 0, NREC, H, H, EPI_NONE);
  }

  // ---- edge softmax: 3 passes (max, normalizer, weighted scatter) ----
  struct EL { int e; const int* ei; int E; const float* q; float* m; float* s; float* agg; };
  EL L[5] = {
    {0, e_ti, NREC, q_item, m_item, s_item, agg_item},
    {1, e_ii, NREC, q_item, m_item, s_item, agg_item},
    {2, e_gi, NREC, q_item, m_item, s_item, agg_item},
    {3, e_ui, EBUY, q_item, m_item, s_item, agg_item},
    {4, e_iu, EBUY, q_user, m_user, s_user, agg_user},
  };
  for (int i = 0; i < 5; i++)
    k_logit<<<(L[i].E + 7) / 8, 256, 0, stream>>>(lb[i], (unsigned*)L[i].m, L[i].q,
                                                  keB[L[i].e], L[i].ei, L[i].ei + L[i].E,
                                                  L[i].E, p_rel, L[i].e);
  for (int i = 0; i < 5; i++)
    k_pass2<<<(L[i].E + 255) / 256, 256, 0, stream>>>(lb[i], (const unsigned*)L[i].m,
                                                      L[i].s, L[i].ei + L[i].E, L[i].E);
  for (int i = 0; i < 5; i++)
    k_pass3<<<(L[i].E + 7) / 8, 256, 0, stream>>>(L[i].agg, lb[i], L[i].s, veB[L[i].e],
                                                  L[i].ei, L[i].ei + L[i].E, L[i].E);

  // ---- output projections with skip blend (write straight into d_out) ----
  k_gelu<<<(NREC * H) / 256, 256, 0, stream>>>(agg_item, NREC * H);
  k_gelu<<<(NUSER * H) / 256, 256, 0, stream>>>(agg_user, NUSER * H);
  gemm(stream, agg_user, H, W_out + 0, H, b_out + 0, out, H,
       user_x, skipA, 0, NUSER, H, H, EPI_SKIP);
  gemm(stream, agg_item, H, W_out + (size_t)1 * H * H, H, b_out + H,
       out + (size_t)NUSER * H, H, item_x_in, skipA, 1, NREC, H, H, EPI_SKIP);

  // ---- concat taste section + broadcast cl_loss ----
  k_final<<<((NUSER + 2 * NREC) * H) / 256, 256, 0, stream>>>(out, taste_x, cl);
}
